// DecoderAR_42863773614113
// MI455X (gfx1250) — matrix operand order repaired and checked by reading, not compiled, not run on hardware
//
#include <hip/hip_runtime.h>

typedef __bf16 bf16_t;
typedef bf16_t v16bf __attribute__((ext_vector_type(16)));
typedef bf16_t v8bf  __attribute__((ext_vector_type(8)));
typedef float  v8f   __attribute__((ext_vector_type(8)));

// explicit global-address-space pointers so laundering keeps global_load_b128
typedef const __attribute__((address_space(1))) bf16_t* gbf16p;
typedef const __attribute__((address_space(1))) v16bf* gv16bfp;

#define B_TOT    8192
#define HORIZON  24
#define NUM_COV  7
#define HID      512
#define GATES    (4*HID)     // 2048
#define KPAD     544         // 512 h + 7 x + 1 y + 24 zero-pad = 17*32
#define KSTEPS   (KPAD/32)   // 17
#define BTILE    32          // 2 WMMA M-tiles -> each B fragment reused twice
#define MTILES   2
#define NWAVES   8
#define THREADS  (NWAVES*32)
#define WC_BYTES ((size_t)GATES*KPAD*2)

// ---------------- prep: fuse weights to bf16, fuse biases ----------------
__global__ __launch_bounds__(256) void prep_kernel(
    const float* __restrict__ W_ih,   // (2048, 8)
    const float* __restrict__ W_hh,   // (2048, 512)
    const float* __restrict__ b_ih,   // (2048,)
    const float* __restrict__ b_hh,   // (2048,)
    bf16_t* __restrict__ Wc,          // (2048, 544) bf16
    float*  __restrict__ bias)        // (2048,)
{
    const int g = blockIdx.x;
    for (int k = threadIdx.x; k < KPAD; k += blockDim.x) {
        float v;
        if (k < HID)            v = W_hh[(size_t)g*HID + k];
        else if (k < HID + 8)   v = W_ih[(size_t)g*8 + (k - HID)];
        else                    v = 0.0f;
        Wc[(size_t)g*KPAD + k] = (bf16_t)v;
    }
    if (threadIdx.x == 0) bias[g] = b_ih[g] + b_hh[g];
}

// ---------------- fast activations (hardware transcendentals) ----------------
#if __has_builtin(__builtin_amdgcn_tanhf)
__device__ __forceinline__ float tanh_fast(float x) {
    return __builtin_amdgcn_tanhf(x);                 // v_tanh_f32
}
__device__ __forceinline__ float sigmoid_fast(float x) {
    return __builtin_fmaf(0.5f, __builtin_amdgcn_tanhf(0.5f * x), 0.5f);
}
#else
#define LOG2E_F 1.44269504088896340736f
__device__ __forceinline__ float sigmoid_fast(float x) {
    return __builtin_amdgcn_rcpf(1.0f + __builtin_amdgcn_exp2f(-LOG2E_F * x));
}
__device__ __forceinline__ float tanh_fast(float x) {
    return __builtin_fmaf(2.0f, sigmoid_fast(2.0f * x), -1.0f);
}
#endif

// ---------------- main persistent decoder ----------------
__global__ __launch_bounds__(THREADS, 1) void decoder_kernel(
    const float* __restrict__ future_x,  // (B, 24, 7)
    const float* __restrict__ h_enc,     // (B, 512)
    const float* __restrict__ c_enc,     // (B, 512)
    const float* __restrict__ y0,        // (B, 1)
    const float* __restrict__ fc_w,      // (1, 512)
    const float* __restrict__ fc_b,      // (1,)
    const bf16_t* __restrict__ Wc,       // (2048, 544) bf16
    const float* __restrict__ bias,      // (2048,)
    float* __restrict__ out)             // (B, 24, 1)
{
    // double-buffered input vector [h | x | y | pad] per row (bf16)
    __shared__ bf16_t lds_in[2][BTILE * KPAD];
    __shared__ float  lds_bias[GATES];
    __shared__ float  lds_fcw[HID];
    __shared__ float  lds_red[BTILE][8];
    __shared__ float  lds_y[BTILE];

    const int tid  = threadIdx.x;
    const int wave = tid >> 5;
    const int lane = tid & 31;
    const int l15  = lane & 15;
    const int hi   = lane >> 4;          // which 16-lane half
    const int r0   = blockIdx.x * BTILE; // batch tile base

    // ---- init LDS ----
    for (int idx = tid; idx < BTILE * HID; idx += THREADS) {
        int row = idx >> 9;
        int k   = idx & (HID - 1);
        lds_in[0][row * KPAD + k] = (bf16_t)h_enc[(size_t)(r0 + row) * HID + k];
    }
    for (int idx = tid; idx < 2 * BTILE * (KPAD - HID); idx += THREADS) {
        int b = idx / (BTILE * (KPAD - HID));
        int r = (idx / (KPAD - HID)) % BTILE;
        int k = HID + idx % (KPAD - HID);
        lds_in[b][r * KPAD + k] = (bf16_t)0.0f;
    }
    for (int k = tid; k < GATES; k += THREADS) lds_bias[k] = bias[k];
    for (int k = tid; k < HID; k += THREADS)   lds_fcw[k]  = fc_w[k];
    if (tid < BTILE) lds_y[tid] = y0[r0 + tid];

    // ---- per-wave recurrent cell state, kept in WMMA C/D fragment layout ----
    const int u_base = wave * 64;        // this wave owns hidden units [u_base, u_base+64)
    v8f c_frag[MTILES][4];
#pragma unroll
    for (int mt = 0; mt < MTILES; ++mt)
#pragma unroll
        for (int gi = 0; gi < 4; ++gi) {
            int u = u_base + gi * 16 + l15;
#pragma unroll
            for (int d = 0; d < 8; ++d)
                c_frag[mt][gi][d] =
                    c_enc[(size_t)(r0 + mt * 16 + d + 8 * hi) * HID + u];
        }

    const int abase = l15 * KPAD + hi * 16;  // A-fragment LDS base, M-tile 0
    const float fcb = fc_b[0];

#pragma unroll 1
    for (int t = 0; t < HORIZON; ++t) {
        bf16_t* cur = lds_in[t & 1];
        bf16_t* nxt = lds_in[(t & 1) ^ 1];

        // ---- stage x_t and y_prev into the input vector ----
        if (tid < BTILE * NUM_COV) {
            int row = tid / NUM_COV;
            int j   = tid % NUM_COV;
            cur[row * KPAD + HID + j] =
                (bf16_t)future_x[((size_t)(r0 + row) * HORIZON + t) * NUM_COV + j];
        }
        if (tid < BTILE)
            cur[tid * KPAD + HID + NUM_COV] = (bf16_t)lds_y[tid];
        __syncthreads();

        // Launder the (t-invariant) weight base so LICM cannot hoist the weight
        // stream out of the recurrence; addrspace(1) keeps global_load_b128.
        gbf16p wc_t = (gbf16p)Wc;
        asm volatile("" : "+s"(wc_t));

        // ---- gates = [h|x|y] @ Wc^T + b via bf16 WMMA; B reused for 2 M-tiles,
        //      B stream software-pipelined one K-step ahead ----
#pragma unroll 1
        for (int gi = 0; gi < 4; ++gi) {
            const int u = u_base + gi * 16;
            const float bI = lds_bias[u + l15];
            const float bF = lds_bias[HID     + u + l15];
            const float bG = lds_bias[2 * HID + u + l15];
            const float bO = lds_bias[3 * HID + u + l15];

            v8f acc[MTILES][4];
#pragma unroll
            for (int mt = 0; mt < MTILES; ++mt)
#pragma unroll
                for (int d = 0; d < 8; ++d) {
                    acc[mt][0][d] = bI; acc[mt][1][d] = bF;
                    acc[mt][2][d] = bG; acc[mt][3][d] = bO;
                }

            gbf16p w0 = wc_t + (size_t)(u + l15) * KPAD;
            const int koff = hi * 16;

            // prologue: B fragments for kb = 0
            v16bf bfr[2][4];
#pragma unroll
            for (int gb = 0; gb < 4; ++gb)
                bfr[0][gb] = *(gv16bfp)(w0 + (size_t)gb * HID * KPAD + koff);

#pragma unroll
            for (int kb = 0; kb < KSTEPS; ++kb) {
                const int cb = kb & 1;
                const int nb = cb ^ 1;

                // prefetch next K-step's B fragments while this step computes
                if (kb + 1 < KSTEPS) {
                    const int kon = (kb + 1) * 32 + koff;
#pragma unroll
                    for (int gb = 0; gb < 4; ++gb)
                        bfr[nb][gb] =
                            *(gv16bfp)(w0 + (size_t)gb * HID * KPAD + kon);
                }

                // two A fragments (rows 0-15 and 16-31), shared across 4 gates
                v16bf a[MTILES];
#pragma unroll
                for (int mt = 0; mt < MTILES; ++mt) {
                    const bf16_t* ap = &cur[abase + mt * 16 * KPAD + kb * 32];
                    v8bf a0 = *(const v8bf*)(ap);
                    v8bf a1 = *(const v8bf*)(ap + 8);
#pragma unroll
                    for (int e = 0; e < 8; ++e) { a[mt][e] = a0[e]; a[mt][e + 8] = a1[e]; }
                }

#pragma unroll
                for (int gb = 0; gb < 4; ++gb) {
                    acc[0][gb] = __builtin_amdgcn_wmma_f32_16x16x32_bf16(
                        false, a[0], false, bfr[cb][gb], (short)0, acc[0][gb], false, false);
                    acc[1][gb] = __builtin_amdgcn_wmma_f32_16x16x32_bf16(
                        false, a[1], false, bfr[cb][gb], (short)0, acc[1][gb], false, false);
                }
            }

            // pointwise LSTM update; h_new goes straight into the next buffer
#pragma unroll
            for (int mt = 0; mt < MTILES; ++mt)
#pragma unroll
                for (int d = 0; d < 8; ++d) {
                    float iv = sigmoid_fast(acc[mt][0][d]);
                    float fv = sigmoid_fast(acc[mt][1][d]);
                    float gv = tanh_fast(acc[mt][2][d]);
                    float ov = sigmoid_fast(acc[mt][3][d]);
                    float cn = fv * c_frag[mt][gi][d] + iv * gv;
                    c_frag[mt][gi][d] = cn;
                    nxt[(mt * 16 + d + 8 * hi) * KPAD + u + l15] =
                        (bf16_t)(ov * tanh_fast(cn));
                }
        }
        __syncthreads();   // h_new complete in nxt

        // ---- fc head: logit = h_new . fc_w + fc_b ; y = sigmoid(logit) ----
        {
            int row = tid >> 3;          // 32 rows
            int seg = tid & 7;           // 8 segments of 64 elements
            float s = 0.0f;
            const v8bf*  hv = (const v8bf*)&nxt[row * KPAD + seg * 64];
            const float* wp = &lds_fcw[seg * 64];
#pragma unroll
            for (int c = 0; c < 8; ++c) {
                v8bf hh = hv[c];
#pragma unroll
                for (int k = 0; k < 8; ++k)
                    s = __builtin_fmaf((float)hh[k], wp[c * 8 + k], s);
            }
            lds_red[row][seg] = s;
        }
        __syncthreads();
        if (tid < BTILE) {
            float s = fcb;
#pragma unroll
            for (int k = 0; k < 8; ++k) s += lds_red[tid][k];
            out[(size_t)(r0 + tid) * HORIZON + t] = s;   // reference returns logits
            lds_y[tid] = sigmoid_fast(s);
        }
        // next iteration's staging barrier orders lds_y/lds_red reuse
    }
}

extern "C" void kernel_launch(void* const* d_in, const int* in_sizes, int n_in,
                              void* d_out, int out_size, void* d_ws, size_t ws_size,
                              hipStream_t stream) {
    const float* future_x = (const float*)d_in[0];
    const float* h_enc    = (const float*)d_in[1];
    const float* c_enc    = (const float*)d_in[2];
    const float* y0       = (const float*)d_in[3];
    const float* W_ih     = (const float*)d_in[4];
    const float* W_hh     = (const float*)d_in[5];
    const float* b_ih     = (const float*)d_in[6];
    const float* b_hh     = (const float*)d_in[7];
    const float* fc_w     = (const float*)d_in[8];
    const float* fc_b     = (const float*)d_in[9];

    bf16_t* Wc   = (bf16_t*)d_ws;
    float*  bias = (float*)((char*)d_ws + WC_BYTES);

    prep_kernel<<<GATES, 256, 0, stream>>>(W_ih, W_hh, b_ih, b_hh, Wc, bias);
    decoder_kernel<<<B_TOT / BTILE, THREADS, 0, stream>>>(
        future_x, h_enc, c_enc, y0, fc_w, fc_b, Wc, bias, (float*)d_out);
}
